// LabelAwareEdgeDecoder_42975442764636
// MI455X (gfx1250) — matrix operand order repaired and checked by reading, not compiled, not run on hardware
//
#include <hip/hip_runtime.h>
#include <math.h>

// ---------------------------------------------------------------------------
// LabelAwareEdgeDecoder: N=768 pairs-MLP (149 -> 256 -> 256 -> 1), bf16 WMMA
// Block = 16 i's x 8 j's = 128 pair rows. Wave w owns col-tiles {2w, 2w+1}
// across all 8 row-tiles, so each W-fragment is fetched from L2 exactly once
// per block (B frags live in registers across the row loop).
// ---------------------------------------------------------------------------

#define Nn    768
#define Dd    32
#define Cc    10
#define Hh    256
#define KPAD  160      // 149 padded up to multiple of 32
#define KT1   5        // 160/32 k-steps, layer 1
#define KT2   8        // 256/32 k-steps, layer 2
#define NT    16       // 256/16 col tiles

typedef __attribute__((ext_vector_type(16))) __bf16         v16bf;
typedef __attribute__((ext_vector_type(8)))  float          v8f;
typedef __attribute__((ext_vector_type(8)))  unsigned short ush8;

union BFrag {
  ush8  h[2];
  v16bf v;
};

__device__ __forceinline__ float bf2f(unsigned short h) {
  union { unsigned u; float f; } x; x.u = ((unsigned)h) << 16;
  return x.f;
}

// ---------------------------------------------------------------------------
// Prep: re-tile W1 (149x256, K-padded to 160) and W2 (256x256) into WMMA
// B-fragment-major bf16 layout: frag(kt,nt) = 32 lanes x 16 contiguous bf16,
// lane = (half, n): element e -> (K = kt*32 + half*16 + e, col = nt*16 + n).
// ---------------------------------------------------------------------------
__global__ __launch_bounds__(256) void prep_kernel(
    const float* __restrict__ W1, const float* __restrict__ W2,
    __bf16* __restrict__ w1f, __bf16* __restrict__ w2f) {
  int idx = blockIdx.x * blockDim.x + threadIdx.x;
  if (idx < KT1 * NT * 512) {
    int e    = idx & 15;
    int lane = (idx >> 4) & 31;
    int frag = idx >> 9;
    int kt = frag >> 4, nt = frag & 15;
    int n = nt * 16 + (lane & 15);
    int k = kt * 32 + (lane >> 4) * 16 + e;
    float v = (k < 149) ? W1[k * Hh + n] : 0.0f;
    w1f[idx] = (__bf16)v;
  }
  int idx2 = idx - KT1 * NT * 512;
  if (idx2 >= 0 && idx2 < KT2 * NT * 512) {
    int e    = idx2 & 15;
    int lane = (idx2 >> 4) & 31;
    int frag = idx2 >> 9;
    int kt = frag >> 4, nt = frag & 15;
    int n = nt * 16 + (lane & 15);
    int k = kt * 32 + (lane >> 4) * 16 + e;
    w2f[idx2] = (__bf16)W2[k * Hh + n];
  }
}

// ---------------------------------------------------------------------------
// Main MLP kernel
// ---------------------------------------------------------------------------
__global__ __launch_bounds__(256) void edge_mlp_kernel(
    const float* __restrict__ z, const float* __restrict__ lp,
    const float* __restrict__ hom,
    const float* __restrict__ b1, const float* __restrict__ b2,
    const float* __restrict__ W3, const float* __restrict__ b3,
    const __bf16* __restrict__ w1f,
    const __bf16* __restrict__ w2f,
    float* __restrict__ P) {
  // Region A (64KB): feat (128x160 bf16 = 40KB) then reused as h2 (128x256 bf16)
  // Region B (64KB): h1 (128x256 bf16)
  __shared__ __align__(16) unsigned char smem[131072];
  __bf16* feat = (__bf16*)smem;
  __bf16* h1   = (__bf16*)(smem + 65536);
  __bf16* h2   = (__bf16*)smem;

  const int i0 = blockIdx.y * 16;
  const int j0 = blockIdx.x * 8;
  const int tid = threadIdx.x;

  // ---- build pair features into LDS (bf16, row-major, K padded to 160) ----
  const float homv = hom[0];
  for (int idx = tid; idx < 128 * KPAD; idx += 256) {
    int p   = idx / KPAD;
    int col = idx - p * KPAD;
    int i = i0 + (p >> 3);
    int j = j0 + (p & 7);
    float v;
    if (col < 32)       v = z[i * Dd + col];
    else if (col < 64)  v = z[j * Dd + (col - 32)];
    else if (col < 96)  { float a = z[i * Dd + (col - 64)], b = z[j * Dd + (col - 64)]; v = fabsf(a - b); }
    else if (col < 128) { float a = z[i * Dd + (col - 96)], b = z[j * Dd + (col - 96)]; v = a * b; }
    else if (col < 138) v = lp[i * Cc + (col - 128)];
    else if (col < 148) v = lp[j * Cc + (col - 138)];
    else if (col == 148) v = homv;
    else                v = 0.0f;
    feat[idx] = (__bf16)v;
  }
  __syncthreads();

  const int wave = tid >> 5;
  const int lane = tid & 31;
  const int half = lane >> 4;
  const int m    = lane & 15;

  // ---- layer 1: H1 = relu(Feat @ W1 + b1), M=128 K=160 N=256 ----
#pragma unroll 1
  for (int t = 0; t < 2; ++t) {
    const int nt = wave * 2 + t;           // this wave's column tile
    BFrag bfr[KT1];
#pragma unroll
    for (int kt = 0; kt < KT1; ++kt) {
      const __bf16* bp = w1f + (kt * NT + nt) * 512 + lane * 16;
      bfr[kt].h[0] = *(const ush8*)bp;
      bfr[kt].h[1] = *(const ush8*)(bp + 8);
    }
    const float bias = b1[nt * 16 + m];
#pragma unroll 1
    for (int rt = 0; rt < 8; ++rt) {
      BFrag af[KT1];
#pragma unroll
      for (int kt = 0; kt < KT1; ++kt) {
        // A layout: lane(half,m): K runs [half*16,+16)
        const __bf16* rp = feat + (rt * 16 + m) * KPAD + kt * 32 + half * 16;
        af[kt].h[0] = *(const ush8*)rp;
        af[kt].h[1] = *(const ush8*)(rp + 8);
      }
      v8f acc = {0.f, 0.f, 0.f, 0.f, 0.f, 0.f, 0.f, 0.f};
#pragma unroll
      for (int kt = 0; kt < KT1; ++kt)
        acc = __builtin_amdgcn_wmma_f32_16x16x32_bf16(
            false, af[kt].v, false, bfr[kt].v, (short)0, acc, false, false);
#pragma unroll
      for (int r = 0; r < 8; ++r) {
        // D layout: element (M = r + 8*half, N = m)
        h1[(rt * 16 + half * 8 + r) * Hh + nt * 16 + m] =
            (__bf16)fmaxf(acc[r] + bias, 0.0f);
      }
    }
  }
  __syncthreads();   // feat fully dead; h1 fully written

  // ---- layer 2: H2 = relu(H1 @ W2 + b2), M=128 K=256 N=256 ----
#pragma unroll 1
  for (int t = 0; t < 2; ++t) {
    const int nt = wave * 2 + t;
    BFrag bfr[KT2];
#pragma unroll
    for (int kt = 0; kt < KT2; ++kt) {
      const __bf16* bp = w2f + (kt * NT + nt) * 512 + lane * 16;
      bfr[kt].h[0] = *(const ush8*)bp;
      bfr[kt].h[1] = *(const ush8*)(bp + 8);
    }
    const float bias = b2[nt * 16 + m];
#pragma unroll 1
    for (int rt = 0; rt < 8; ++rt) {
      BFrag af[KT2];
#pragma unroll
      for (int kt = 0; kt < KT2; ++kt) {
        const __bf16* rp = h1 + (rt * 16 + m) * Hh + kt * 32 + half * 16;
        af[kt].h[0] = *(const ush8*)rp;
        af[kt].h[1] = *(const ush8*)(rp + 8);
      }
      v8f acc = {0.f, 0.f, 0.f, 0.f, 0.f, 0.f, 0.f, 0.f};
#pragma unroll
      for (int kt = 0; kt < KT2; ++kt)
        acc = __builtin_amdgcn_wmma_f32_16x16x32_bf16(
            false, af[kt].v, false, bfr[kt].v, (short)0, acc, false, false);
#pragma unroll
      for (int r = 0; r < 8; ++r) {
        h2[(rt * 16 + half * 8 + r) * Hh + nt * 16 + m] =
            (__bf16)fmaxf(acc[r] + bias, 0.0f);
      }
    }
  }
  __syncthreads();   // h2 fully written

  // ---- layer 3: logits = H2 @ W3 + b3 ; sigmoid ; zero diagonal ----
  {
    int p  = tid >> 1;        // pair row 0..127
    int hh = tid & 1;         // pair of lanes splits the 256-dot
    const __bf16* rowp = h2 + p * Hh + hh * 128;
    const float* w3p = W3 + hh * 128;
    float s = 0.f;
#pragma unroll 1
    for (int c = 0; c < 128; c += 8) {
      ush8 v = *(const ush8*)(rowp + c);
#pragma unroll
      for (int e = 0; e < 8; ++e) s += bf2f(v[e]) * w3p[c + e];
    }
    float other = __shfl_xor(s, 1, 32);
    if (hh == 0) {
      float logit = s + other + b3[0];
      float pr = 1.0f / (1.0f + expf(-logit));
      int i = i0 + (p >> 3);
      int j = j0 + (p & 7);
      P[i * Nn + j] = (i == j) ? 0.0f : pr;
    }
  }
}

// ---------------------------------------------------------------------------
// Symmetrize: out = 0.5 * (P + P^T)
// ---------------------------------------------------------------------------
__global__ __launch_bounds__(256) void sym_kernel(
    const float* __restrict__ P, float* __restrict__ out) {
  int idx = blockIdx.x * blockDim.x + threadIdx.x;
  if (idx < Nn * Nn) {
    int i = idx / Nn;
    int j = idx - i * Nn;
    out[idx] = 0.5f * (P[idx] + P[j * Nn + i]);
  }
}

extern "C" void kernel_launch(void* const* d_in, const int* in_sizes, int n_in,
                              void* d_out, int out_size, void* d_ws, size_t ws_size,
                              hipStream_t stream) {
  (void)in_sizes; (void)n_in; (void)out_size; (void)ws_size;
  const float* z   = (const float*)d_in[0];
  const float* lp  = (const float*)d_in[1];
  const float* hom = (const float*)d_in[2];
  const float* W1  = (const float*)d_in[3];
  const float* b1  = (const float*)d_in[4];
  const float* W2  = (const float*)d_in[5];
  const float* b2  = (const float*)d_in[6];
  const float* W3  = (const float*)d_in[7];
  const float* b3  = (const float*)d_in[8];

  // workspace layout: [w1f bf16 40960][w2f bf16 65536][P f32 768*768]
  __bf16* w1f = (__bf16*)d_ws;
  __bf16* w2f = (__bf16*)((char*)d_ws + 81920);
  float*  P   = (float*)((char*)d_ws + 81920 + 131072);
  float*  out = (float*)d_out;

  // re-tile weights (tiny: 106496 threads)
  prep_kernel<<<(KT1 * NT * 512 + KT2 * NT * 512 + 255) / 256, 256, 0, stream>>>(
      W1, W2, w1f, w2f);

  // main MLP over all pair tiles: 96 j-tiles x 48 i-tiles
  dim3 grid(Nn / 8, Nn / 16);
  edge_mlp_kernel<<<grid, 256, 0, stream>>>(z, lp, hom, b1, b2, W3, b3, w1f, w2f, P);

  // symmetrize
  sym_kernel<<<(Nn * Nn + 255) / 256, 256, 0, stream>>>(P, out);
}
